// net_7_28587302322777
// MI455X (gfx1250) — matrix operand order repaired and checked by reading, not compiled, not run on hardware
//
#include <hip/hip_runtime.h>

typedef __attribute__((ext_vector_type(16))) __bf16 v16bf;
typedef __attribute__((ext_vector_type(8)))  float  v8f;

union BFrag {
  v16bf v;
  unsigned short h[16];
  uint4 q[2];
};

__device__ __forceinline__ unsigned short f2bf(float f) {
  union { float f; unsigned u; } x; x.f = f;
  unsigned r = x.u + 0x7FFFu + ((x.u >> 16) & 1u);
  return (unsigned short)(r >> 16);
}

__device__ __forceinline__ unsigned pack2bf(float a, float b) {
  return (unsigned)f2bf(a) | ((unsigned)f2bf(b) << 16);
}

// ---------------------------------------------------------------------------
// Prep: convert conv weights to bf16 (K-major rows of 576), pack wq/wk/wv as
// [proj][head][d] rows of 64 channels (q pre-scaled by d^-0.5 = 0.25),
// zero the BN statistics accumulators.
// ---------------------------------------------------------------------------
__global__ void k_prep(const float* __restrict__ w1, const float* __restrict__ w3,
                       const float* __restrict__ wq, const float* __restrict__ wk,
                       const float* __restrict__ wv,
                       unsigned short* __restrict__ w1bf,
                       unsigned short* __restrict__ w3bf,
                       unsigned short* __restrict__ wqkvbf,
                       float* __restrict__ stats) {
  int i = blockIdx.x * blockDim.x + threadIdx.x;
  if (i < 36864)  w1bf[i] = f2bf(w1[i]);
  if (i < 147456) w3bf[i] = f2bf(w3[i]);
  if (i < 12288) {
    int r = i >> 6, c = i & 63;          // r = proj*64 + head*16 + d
    int proj = r >> 6, hd = r & 63;
    const float* src = (proj == 0) ? wq : ((proj == 1) ? wk : wv);
    float v = src[c * 64 + hd];
    if (proj == 0) v *= 0.25f;           // fold q scaling (16^-0.5)
    wqkvbf[i] = f2bf(v);
  }
  if (i < 768) stats[i] = 0.f;
}

// ---------------------------------------------------------------------------
// Build im2col (bf16, paired dword stores) for one 4x4/64ch block held in
// LDS as xblk[1024] (fp32, [c][p]).  im2 row stride = 576 (= 18 WMMA k-steps).
// ---------------------------------------------------------------------------
__device__ __forceinline__ void build_im2col(const float* __restrict__ xblk,
                                             unsigned* __restrict__ im2u,
                                             int tid, int nthreads) {
  // 16*576/2 = 4608 dword pairs per block
  for (int pe = tid; pe < 4608; pe += nthreads) {
    int e = pe * 2;
    int p = e / 576, kk = e - p * 576;
    int py = p >> 2, px = p & 3;
    float v0, v1;
    {
      int ci = kk / 9, t = kk - ci * 9;
      int yy = py + t / 3 - 1, xx = px + (t % 3) - 1;
      v0 = ((unsigned)yy < 4u && (unsigned)xx < 4u) ? xblk[ci * 16 + yy * 4 + xx] : 0.f;
    }
    {
      int k1 = kk + 1;
      int ci = k1 / 9, t = k1 - ci * 9;
      int yy = py + t / 3 - 1, xx = px + (t % 3) - 1;
      v1 = ((unsigned)yy < 4u && (unsigned)xx < 4u) ? xblk[ci * 16 + yy * 4 + xx] : 0.f;
    }
    im2u[pe] = pack2bf(v0, v1);
  }
}

// ---------------------------------------------------------------------------
// conv1: 3x3 conv 64->64 as im2col GEMM (M=16, K=576, N=64).
// TWO blocks per workgroup so each B (weight) fragment feeds two WMMAs,
// halving L2 weight traffic. 4 waves, one 16-wide N tile each.
// ---------------------------------------------------------------------------
__global__ __launch_bounds__(128) void k_conv1(
    const float* __restrict__ x, const unsigned short* __restrict__ wbf,
    const float* __restrict__ bias, float* __restrict__ y,
    float* __restrict__ ssum, float* __restrict__ ssq) {
  __shared__ float xblk[2048];
  __shared__ unsigned short im2[2 * 9216];
  const int b0 = blockIdx.x * 2;
  const int tid = threadIdx.x;
  const float4* xp = (const float4*)(x + (size_t)b0 * 1024);
  for (int i = tid; i < 512; i += 128) ((float4*)xblk)[i] = xp[i];
  __syncthreads();
  build_im2col(xblk,        (unsigned*)im2,        tid, 128);
  build_im2col(xblk + 1024, (unsigned*)(im2 + 9216), tid, 128);
  __syncthreads();
  const int lane = tid & 31, wave = tid >> 5;
  const int sub = lane >> 4, l = lane & 15;
  const int co = wave * 16 + l;
  v8f z8 = {0.f, 0.f, 0.f, 0.f, 0.f, 0.f, 0.f, 0.f};
  v8f acc0 = z8, acc1 = z8;
  const unsigned short* brow  = wbf + co * 576 + sub * 8;
  const unsigned short* arow0 = im2 + l * 576 + sub * 8;
  const unsigned short* arow1 = arow0 + 9216;
  for (int ks = 0; ks < 18; ++ks) {
    BFrag a0, a1, bb;
    a0.q[0] = *(const uint4*)(arow0 + ks * 32);
    a0.q[1] = *(const uint4*)(arow0 + ks * 32 + 16);
    a1.q[0] = *(const uint4*)(arow1 + ks * 32);
    a1.q[1] = *(const uint4*)(arow1 + ks * 32 + 16);
    bb.q[0] = *(const uint4*)(brow + ks * 32);
    bb.q[1] = *(const uint4*)(brow + ks * 32 + 16);
    acc0 = __builtin_amdgcn_wmma_f32_16x16x32_bf16(false, a0.v, false, bb.v,
                                                   (short)0, acc0, false, false);
    acc1 = __builtin_amdgcn_wmma_f32_16x16x32_bf16(false, a1.v, false, bb.v,
                                                   (short)0, acc1, false, false);
  }
  const float bv = bias[co];
  #pragma unroll
  for (int blk = 0; blk < 2; ++blk) {
    const v8f& acc = blk ? acc1 : acc0;
    float t[8], s = 0.f, s2 = 0.f;
    #pragma unroll
    for (int r = 0; r < 8; ++r) { t[r] = acc[r] + bv; s += t[r]; s2 += t[r] * t[r]; }
    float* op = y + (size_t)(b0 + blk) * 1024 + co * 16 + sub * 8;
    *(float4*)op       = make_float4(t[0], t[1], t[2], t[3]);
    *(float4*)(op + 4) = make_float4(t[4], t[5], t[6], t[7]);
    atomicAdd(&ssum[co], s);
    atomicAdd(&ssq[co], s2);
  }
}

// ---------------------------------------------------------------------------
// Finalize BN: scale = g*rsqrt(var+eps), shift = b - mean*scale
// ---------------------------------------------------------------------------
__global__ void k_finalize(const float* __restrict__ ssum, const float* __restrict__ ssq,
                           const float* __restrict__ g, const float* __restrict__ bta,
                           float* __restrict__ sc, float* __restrict__ sh,
                           int C, float invcnt) {
  int c = blockIdx.x * blockDim.x + threadIdx.x;
  if (c < C) {
    float m = ssum[c] * invcnt;
    float var = ssq[c] * invcnt - m * m;
    float s = g[c] * rsqrtf(var + 1e-5f);
    sc[c] = s;
    sh[c] = bta[c] - m * s;
  }
}

// ---------------------------------------------------------------------------
// Attention: BN1+ReLU on the fly, qkv projection WMMAs, QK^T WMMA,
// rel-pos logits + softmax (scalar, per query lane), P*V WMMA.
// One wave per head. Accumulate BN2 stats.
// ---------------------------------------------------------------------------
__global__ __launch_bounds__(128) void k_attn(
    const float* __restrict__ y1, const unsigned short* __restrict__ wqkv,
    const float* __restrict__ relw, const float* __restrict__ relh,
    const float* __restrict__ sc, const float* __restrict__ shv,
    float* __restrict__ o, float* __restrict__ ssum, float* __restrict__ ssq) {
  __shared__ unsigned short feat[16 * 64];   // [p][c] bf16
  __shared__ float qf[4][256];               // [head][p][d]
  __shared__ float kf[4][256];               // [head][p][d]
  __shared__ float vt[4][256];               // [head][d][p']  (transposed)
  __shared__ float lf[4][256];               // [head][p][p']
  __shared__ unsigned short wl[4][256];      // softmax weights bf16
  __shared__ float rw[112], rh[112];
  const int b = blockIdx.x, tid = threadIdx.x;
  for (int i = tid; i < 112; i += 128) { rw[i] = relw[i]; rh[i] = relh[i]; }
  const float* yp = y1 + (size_t)b * 1024;
  for (int i = tid; i < 1024; i += 128) {
    int c = i >> 4, p = i & 15;
    float v = fmaxf(yp[i] * sc[c] + shv[c], 0.f);
    feat[p * 64 + c] = f2bf(v);
  }
  __syncthreads();
  const int lane = tid & 31, hh = tid >> 5;   // hh = head
  const int sub = lane >> 4, l = lane & 15;
  v8f zero8 = {0.f, 0.f, 0.f, 0.f, 0.f, 0.f, 0.f, 0.f};
  v8f aq = zero8, ak = zero8, av = zero8;
  #pragma unroll
  for (int ks = 0; ks < 2; ++ks) {
    BFrag a;
    const unsigned short* ap = feat + l * 64 + ks * 32 + sub * 8;
    a.q[0] = *(const uint4*)ap;
    a.q[1] = *(const uint4*)(ap + 16);
    #pragma unroll
    for (int proj = 0; proj < 3; ++proj) {
      BFrag bb;
      const unsigned short* bp = wqkv + (proj * 64 + hh * 16 + l) * 64 + ks * 32 + sub * 8;
      bb.q[0] = *(const uint4*)bp;
      bb.q[1] = *(const uint4*)(bp + 16);
      if (proj == 0)      aq = __builtin_amdgcn_wmma_f32_16x16x32_bf16(false, a.v, false, bb.v, (short)0, aq, false, false);
      else if (proj == 1) ak = __builtin_amdgcn_wmma_f32_16x16x32_bf16(false, a.v, false, bb.v, (short)0, ak, false, false);
      else                av = __builtin_amdgcn_wmma_f32_16x16x32_bf16(false, a.v, false, bb.v, (short)0, av, false, false);
    }
  }
  #pragma unroll
  for (int r = 0; r < 8; ++r) {
    int p = r + sub * 8;
    qf[hh][p * 16 + l] = aq[r];
    kf[hh][p * 16 + l] = ak[r];
    vt[hh][l * 16 + p] = av[r];
  }
  // QK^T: M=p, N=p', K=d (16 padded to 32)
  BFrag a2, b2;
  {
    const float* qr = &qf[hh][l * 16 + sub * 8];
    #pragma unroll
    for (int j = 0; j < 8; ++j) a2.h[j] = f2bf(qr[j]);
    a2.q[1] = make_uint4(0, 0, 0, 0);
  }
  {
    const float* kr = &kf[hh][l * 16 + sub * 8];
    #pragma unroll
    for (int j = 0; j < 8; ++j) b2.h[j] = f2bf(kr[j]);
    b2.q[1] = make_uint4(0, 0, 0, 0);
  }
  v8f lg = __builtin_amdgcn_wmma_f32_16x16x32_bf16(false, a2.v, false, b2.v,
                                                   (short)0, zero8, false, false);
  #pragma unroll
  for (int r = 0; r < 8; ++r) lf[hh][(r + sub * 8) * 16 + l] = lg[r];

  // softmax + relative logits: lanes 0-15, one query each
  if (sub == 0) {
    const int p = l, qh = p >> 2, qw = p & 3;
    float qrow[16];
    #pragma unroll
    for (int d = 0; d < 16; ++d) qrow[d] = qf[hh][p * 16 + d];
    float RW[7], RH[7];
    for (int m = 0; m < 7; ++m) {
      float sw = 0.f, s2 = 0.f;
      #pragma unroll
      for (int d = 0; d < 16; ++d) { sw += qrow[d] * rw[m * 16 + d]; s2 += qrow[d] * rh[m * 16 + d]; }
      RW[m] = sw; RH[m] = s2;
    }
    float lv[16]; float mx = -3.4e38f;
    #pragma unroll
    for (int kk = 0; kk < 16; ++kk) {
      int kh = kk >> 2, kw = kk & 3;
      float v = lf[hh][p * 16 + kk] + RW[kw - qw + 3] + RH[kh - qh + 3];
      lv[kk] = v; mx = fmaxf(mx, v);
    }
    float sm = 0.f;
    #pragma unroll
    for (int kk = 0; kk < 16; ++kk) { float e = __expf(lv[kk] - mx); lv[kk] = e; sm += e; }
    float inv = 1.f / sm;
    #pragma unroll
    for (int kk = 0; kk < 16; ++kk) wl[hh][p * 16 + kk] = f2bf(lv[kk] * inv);
  }

  // P*V: M=p, N=d, K=p' (16 padded to 32)
  BFrag a3, b3;
  a3.q[0] = *(const uint4*)(&wl[hh][l * 16 + sub * 8]);
  a3.q[1] = make_uint4(0, 0, 0, 0);
  {
    const float* vr = &vt[hh][l * 16 + sub * 8];
    #pragma unroll
    for (int j = 0; j < 8; ++j) b3.h[j] = f2bf(vr[j]);
    b3.q[1] = make_uint4(0, 0, 0, 0);
  }
  v8f outf = __builtin_amdgcn_wmma_f32_16x16x32_bf16(false, a3.v, false, b3.v,
                                                     (short)0, zero8, false, false);
  const int co = hh * 16 + l;
  float t[8], s = 0.f, s2 = 0.f;
  #pragma unroll
  for (int r = 0; r < 8; ++r) { t[r] = outf[r]; s += t[r]; s2 += t[r] * t[r]; }
  float* op = o + (size_t)b * 1024 + co * 16 + sub * 8;
  *(float4*)op       = make_float4(t[0], t[1], t[2], t[3]);
  *(float4*)(op + 4) = make_float4(t[4], t[5], t[6], t[7]);
  atomicAdd(&ssum[co], s);
  atomicAdd(&ssq[co], s2);
}

// ---------------------------------------------------------------------------
// conv3: BN2+ReLU on the fly, 3x3 conv 64->256 (M=16, K=576, N=256).
// TWO blocks per workgroup; 4 waves x 4 N-tiles, each B fragment feeds two
// WMMAs. Raw conv output -> d_out; BN3 stats accumulated.
// ---------------------------------------------------------------------------
__global__ __launch_bounds__(128) void k_conv3(
    const float* __restrict__ oin, const unsigned short* __restrict__ wbf,
    const float* __restrict__ bias, const float* __restrict__ sc,
    const float* __restrict__ shv, float* __restrict__ y,
    float* __restrict__ ssum, float* __restrict__ ssq) {
  __shared__ float xblk[2048];
  __shared__ unsigned short im2[2 * 9216];
  const int b0 = blockIdx.x * 2;
  const int tid = threadIdx.x;
  const float4* xp = (const float4*)(oin + (size_t)b0 * 1024);
  for (int i = tid; i < 512; i += 128) {
    float4 v = xp[i];
    int c = ((i * 4) & 1023) >> 4;        // same channel for all 4 lanes of the vec
    float s = sc[c], b = shv[c];
    v.x = fmaxf(v.x * s + b, 0.f);
    v.y = fmaxf(v.y * s + b, 0.f);
    v.z = fmaxf(v.z * s + b, 0.f);
    v.w = fmaxf(v.w * s + b, 0.f);
    ((float4*)xblk)[i] = v;
  }
  __syncthreads();
  build_im2col(xblk,        (unsigned*)im2,          tid, 128);
  build_im2col(xblk + 1024, (unsigned*)(im2 + 9216), tid, 128);
  __syncthreads();
  const int lane = tid & 31, wave = tid >> 5;
  const int sub = lane >> 4, l = lane & 15;
  const unsigned short* arow0 = im2 + l * 576 + sub * 8;
  const unsigned short* arow1 = arow0 + 9216;
  const unsigned short* brow  = wbf + (wave * 64 + l) * 576 + sub * 8;
  v8f z8 = {0.f, 0.f, 0.f, 0.f, 0.f, 0.f, 0.f, 0.f};
  v8f acc[8];
  #pragma unroll
  for (int j = 0; j < 8; ++j) acc[j] = z8;
  for (int ks = 0; ks < 18; ++ks) {
    BFrag a0, a1;
    a0.q[0] = *(const uint4*)(arow0 + ks * 32);
    a0.q[1] = *(const uint4*)(arow0 + ks * 32 + 16);
    a1.q[0] = *(const uint4*)(arow1 + ks * 32);
    a1.q[1] = *(const uint4*)(arow1 + ks * 32 + 16);
    #pragma unroll
    for (int j = 0; j < 4; ++j) {
      BFrag bb;
      const unsigned short* bp = brow + j * 9216 + ks * 32;  // j*16 rows * 576
      bb.q[0] = *(const uint4*)bp;
      bb.q[1] = *(const uint4*)(bp + 16);
      acc[j * 2]     = __builtin_amdgcn_wmma_f32_16x16x32_bf16(false, a0.v, false, bb.v,
                                                               (short)0, acc[j * 2], false, false);
      acc[j * 2 + 1] = __builtin_amdgcn_wmma_f32_16x16x32_bf16(false, a1.v, false, bb.v,
                                                               (short)0, acc[j * 2 + 1], false, false);
    }
  }
  #pragma unroll
  for (int j = 0; j < 4; ++j) {
    const int co = wave * 64 + j * 16 + l;
    const float bv = bias[co];
    #pragma unroll
    for (int blk = 0; blk < 2; ++blk) {
      float t[8], s = 0.f, s2 = 0.f;
      #pragma unroll
      for (int r = 0; r < 8; ++r) {
        t[r] = acc[j * 2 + blk][r] + bv; s += t[r]; s2 += t[r] * t[r];
      }
      float* op = y + (size_t)(b0 + blk) * 4096 + co * 16 + sub * 8;
      *(float4*)op       = make_float4(t[0], t[1], t[2], t[3]);
      *(float4*)(op + 4) = make_float4(t[4], t[5], t[6], t[7]);
      atomicAdd(&ssum[co], s);
      atomicAdd(&ssq[co], s2);
    }
  }
}

// ---------------------------------------------------------------------------
// Final elementwise BN3 + ReLU in place on d_out.
// ---------------------------------------------------------------------------
__global__ void k_bnrelu(float* __restrict__ y, const float* __restrict__ sc,
                         const float* __restrict__ shv, int n4) {
  int i = blockIdx.x * blockDim.x + threadIdx.x;
  if (i >= n4) return;
  int idx = i * 4;
  int c = (idx >> 4) & 255;
  float s = sc[c], bsh = shv[c];
  float4 v = *(float4*)(y + idx);
  v.x = fmaxf(v.x * s + bsh, 0.f);
  v.y = fmaxf(v.y * s + bsh, 0.f);
  v.z = fmaxf(v.z * s + bsh, 0.f);
  v.w = fmaxf(v.w * s + bsh, 0.f);
  *(float4*)(y + idx) = v;
}

extern "C" void kernel_launch(void* const* d_in, const int* in_sizes, int n_in,
                              void* d_out, int out_size, void* d_ws, size_t ws_size,
                              hipStream_t stream) {
  (void)n_in; (void)ws_size;
  const float* x      = (const float*)d_in[0];
  const float* conv1w = (const float*)d_in[1];
  const float* conv1b = (const float*)d_in[2];
  const float* bn1g   = (const float*)d_in[3];
  const float* bn1b   = (const float*)d_in[4];
  const float* wq     = (const float*)d_in[5];
  const float* wk     = (const float*)d_in[6];
  const float* wv     = (const float*)d_in[7];
  const float* relw   = (const float*)d_in[8];
  const float* relh   = (const float*)d_in[9];
  const float* bn2g   = (const float*)d_in[10];
  const float* bn2b   = (const float*)d_in[11];
  const float* conv3w = (const float*)d_in[12];
  const float* conv3b = (const float*)d_in[13];
  const float* bn3g   = (const float*)d_in[14];
  const float* bn3b   = (const float*)d_in[15];
  float* out = (float*)d_out;

  const int NBLK = in_sizes[0] / 1024;              // 8192 4x4 blocks
  const float invcnt = 1.0f / (float)(NBLK * 16);   // BN count = N*H*W

  char* ws = (char*)d_ws;
  size_t off = 0;
  float* y1    = (float*)(ws + off); off += (size_t)NBLK * 1024 * 4;
  float* oattn = (float*)(ws + off); off += (size_t)NBLK * 1024 * 4;
  unsigned short* w1bf   = (unsigned short*)(ws + off); off += 64 * 576 * 2;
  unsigned short* w3bf   = (unsigned short*)(ws + off); off += 256 * 576 * 2;
  unsigned short* wqkvbf = (unsigned short*)(ws + off); off += 192 * 64 * 2;
  float* stats = (float*)(ws + off); off += 768 * 4;   // s1sum s1sq s2sum s2sq s3sum s3sq
  float* scsh  = (float*)(ws + off); off += 768 * 4;   // sc1 sh1 sc2 sh2 sc3 sh3

  k_prep<<<576, 256, 0, stream>>>(conv1w, conv3w, wq, wk, wv, w1bf, w3bf, wqkvbf, stats);

  k_conv1<<<NBLK / 2, 128, 0, stream>>>(x, w1bf, conv1b, y1, stats + 0, stats + 64);
  k_finalize<<<1, 64, 0, stream>>>(stats + 0, stats + 64, bn1g, bn1b,
                                   scsh + 0, scsh + 64, 64, invcnt);

  k_attn<<<NBLK, 128, 0, stream>>>(y1, wqkvbf, relw, relh, scsh + 0, scsh + 64,
                                   oattn, stats + 128, stats + 192);
  k_finalize<<<1, 64, 0, stream>>>(stats + 128, stats + 192, bn2g, bn2b,
                                   scsh + 128, scsh + 192, 64, invcnt);

  k_conv3<<<NBLK / 2, 128, 0, stream>>>(oattn, w3bf, conv3b, scsh + 128, scsh + 192,
                                        out, stats + 256, stats + 512);
  k_finalize<<<4, 64, 0, stream>>>(stats + 256, stats + 512, bn3g, bn3b,
                                   scsh + 256, scsh + 512, 256, invcnt);

  k_bnrelu<<<out_size / 1024, 256, 0, stream>>>(out, scsh + 256, scsh + 512,
                                                out_size / 4);
}
